// LEDDecoderAttentionTopicAware_35235911696716
// MI455X (gfx1250) — matrix operand order repaired and checked by reading, not compiled, not run on hardware
//
#include <hip/hip_runtime.h>

// LEDDecoderAttentionTopicAware for gfx1250 (MI455X): bf16 WMMA everywhere.
// Pipeline: cvt inputs->bf16, transpose weights->bf16, WMMA GEMM projections
// (q scaled, k, v^T, kt, vt^T in head-major layouts), flash-style streaming
// attention (cooperative: balanced score WMMAs -> LDS softmax -> shared-P PV
// WMMAs; zero redundant matrix ops, fragment loads prefetched ahead of the
// WMMA chains), gate GEMV + sigmoid, elementwise mix, final WMMA GEMM @ Wo.

#define DEVI __device__ __forceinline__

typedef __bf16 bf16_t;
typedef __bf16 v16bf __attribute__((ext_vector_type(16)));
typedef float  v8f   __attribute__((ext_vector_type(8)));

static constexpr int Bb = 2, T = 512, S = 4096, E = 1024, TP = 512, H = 8, DH = 128;

DEVI v8f vzero() {
  v8f z;
#pragma unroll
  for (int i = 0; i < 8; ++i) z[i] = 0.0f;
  return z;
}

DEVI v8f wmma_bf16(v16bf a, v16bf b, v8f c) {
  return __builtin_amdgcn_wmma_f32_16x16x32_bf16(false, a, false, b, (short)0, c, false, false);
}

// A-fragment (16x32, M=lane%16): elem e = K[half*16 + e]
DEVI v16bf load_a16(const bf16_t* rowp, int half) {
  union { uint4 u[2]; v16bf v; } u;
  u.u[0] = *(const uint4*)(rowp + half * 16);
  u.u[1] = *(const uint4*)(rowp + half * 16 + 8);
  return u.v;
}

// B-fragment (32x16, N=lane%16): elem e = K[half*16 + e]; caller passes p = &B[N][half*16]
DEVI v16bf load_b16(const bf16_t* p) {
  union { uint4 u[2]; v16bf v; } u;
  u.u[0] = *(const uint4*)(p);
  u.u[1] = *(const uint4*)(p + 8);
  return u.v;
}

// ---------------- conversion / transpose ----------------
__global__ void cvt_k(const float* __restrict__ src, bf16_t* __restrict__ dst, size_t n) {
  size_t i = (size_t)blockIdx.x * 256 + threadIdx.x;
  if (i < n) dst[i] = (bf16_t)src[i];
}

// src [K,N] f32 (torch-style y = x @ W) -> dst [N,K] bf16
__global__ void trans_k(const float* __restrict__ src, bf16_t* __restrict__ dst, int K, int N) {
  size_t i = (size_t)blockIdx.x * 256 + threadIdx.x;
  if (i < (size_t)K * N) {
    int n = (int)(i / K), k = (int)(i % K);
    dst[i] = (bf16_t)src[(size_t)k * N + n];
  }
}

// ---------------- GEMM: out = (A @ BT^T + bias) * scale ----------------
// MODE 0: plain [M,N]; MODE 1: head-major [B,H,L,DH]; MODE 2: head-major transposed [B,H,DH,L]
template <int MODE, bool OUTBF>
__global__ __launch_bounds__(128) void gemm_k(const bf16_t* __restrict__ A,
                                              const bf16_t* __restrict__ BT,
                                              const float* __restrict__ bias,
                                              void* __restrict__ out,
                                              int M, int N, int K, int L, float scale) {
  __shared__ __align__(16) bf16_t As[64][40];
  __shared__ __align__(16) bf16_t Bs[64][40];
  const int tid = threadIdx.x;
  const int lane = tid & 31, wave = tid >> 5;
  const int half = lane >> 4, ln = lane & 15;
  const int wm = (wave >> 1) * 32, wn = (wave & 1) * 32;
  const int m0 = blockIdx.y * 64, n0 = blockIdx.x * 64;

  v8f c[2][2];
#pragma unroll
  for (int i = 0; i < 2; ++i)
#pragma unroll
    for (int j = 0; j < 2; ++j) c[i][j] = vzero();

  for (int kc = 0; kc < K; kc += 32) {
    for (int u = tid; u < 256; u += 128) {
      int row = u >> 2, seg = u & 3;
      *(uint4*)&As[row][seg * 8] = *(const uint4*)&A[(size_t)(m0 + row) * K + kc + seg * 8];
      *(uint4*)&Bs[row][seg * 8] = *(const uint4*)&BT[(size_t)(n0 + row) * K + kc + seg * 8];
    }
    __syncthreads();
    v16bf a0 = load_a16(&As[wm + ln][0], half);
    v16bf a1 = load_a16(&As[wm + 16 + ln][0], half);
    v16bf b0 = load_b16(&Bs[wn + ln][half * 16]);
    v16bf b1 = load_b16(&Bs[wn + 16 + ln][half * 16]);
    c[0][0] = wmma_bf16(a0, b0, c[0][0]);
    c[0][1] = wmma_bf16(a0, b1, c[0][1]);
    c[1][0] = wmma_bf16(a1, b0, c[1][0]);
    c[1][1] = wmma_bf16(a1, b1, c[1][1]);
    __syncthreads();
  }

#pragma unroll
  for (int i = 0; i < 2; ++i)
#pragma unroll
    for (int j = 0; j < 2; ++j)
#pragma unroll
      for (int r = 0; r < 8; ++r) {
        int row = m0 + wm + i * 16 + r + 8 * half;
        int col = n0 + wn + j * 16 + ln;
        float v = (c[i][j][r] + bias[col]) * scale;
        size_t off;
        if (MODE == 0) {
          off = (size_t)row * N + col;
        } else {
          int bb = row / L, l = row % L;
          int h = col >> 7, d = col & 127;
          if (MODE == 1) off = (((size_t)(bb * H + h)) * L + l) * DH + d;
          else off = (((size_t)(bb * H + h)) * DH + d) * (size_t)L + l;
        }
        if constexpr (OUTBF) ((bf16_t*)out)[off] = (bf16_t)v;
        else ((float*)out)[off] = v;
      }
}

// ---------------- cooperative streaming dual-softmax attention ----------------
// grid.x = B*H * (T/32); block = 256 (8 waves).
// Phase 1: 8 score tiles (2 rowgroups x 2 types x 2 coltiles) -> one per wave (4 WMMAs each).
// Phase 2: waves 0-3 run online softmax for one (rowgroup,type) combo; P + alpha to LDS.
// Phase 3: all waves (rowgroup x dh-quarter) rescale accumulators and do 6 PV WMMAs.
__global__ __launch_bounds__(256) void attn_k(const bf16_t* __restrict__ qh,
                                              const bf16_t* __restrict__ kh,
                                              const bf16_t* __restrict__ kth,
                                              const bf16_t* __restrict__ vTb,
                                              const bf16_t* __restrict__ vtTb,
                                              float* __restrict__ ctx,
                                              float* __restrict__ ctxT,
                                              float* __restrict__ ctxTV) {
  __shared__ __align__(16) float ldsS[2][2][2][16][20];  // [rg][type][coltile][row][col(+pad)]
  __shared__ __align__(16) bf16_t ldsP[2][2][16 * 40];   // [rg][type][row*40 + col]
  __shared__ float alphaL[2][2][16];
  __shared__ float mstate[2][2][16];
  __shared__ float sstate[2][2][16];

  const int tid = threadIdx.x;
  const int lane = tid & 31;
  const int wave = __builtin_amdgcn_readfirstlane(tid >> 5);  // wave-uniform SGPR role
  const int half = lane >> 4, ln = lane & 15;
  // score-phase role
  const int rg_s = wave & 1, ty_s = (wave >> 1) & 1, hs = wave >> 2;
  // pv-phase role
  const int rg_p = wave >> 2, dq = wave & 3;
  const int bh = blockIdx.x >> 4;
  const int trow0 = (blockIdx.x & 15) * 32;

  if (tid < 64) {
    int rg = tid >> 5, ty = (tid >> 4) & 1, row = tid & 15;
    mstate[rg][ty][row] = -__builtin_inff();
    sstate[rg][ty][row] = 0.f;
  }

  // q rows for this wave's score tile
  const bf16_t* qbase = qh + ((size_t)bh * T + trow0 + rg_s * 16 + ln) * DH;
  v16bf qa[4];
#pragma unroll
  for (int kc = 0; kc < 4; ++kc) qa[kc] = load_a16(qbase + kc * 32, half);

  const bf16_t* kb = (ty_s == 0 ? kh : kth) + (size_t)bh * S * DH + (size_t)(hs * 16 + ln) * DH;
  const bf16_t* vb = vTb + (size_t)bh * DH * S;
  const bf16_t* vtb = vtTb + (size_t)bh * DH * S;

  v8f accR[2], accT[2], accTV[2];
#pragma unroll
  for (int g = 0; g < 2; ++g) { accR[g] = vzero(); accT[g] = vzero(); accTV[g] = vzero(); }

  __syncthreads();

  for (int s0 = 0; s0 < S; s0 += 32) {
    // ---- phase 1: one 16x16 score tile per wave; prefetch all 4 B-fragments ----
    const bf16_t* kp = kb + (size_t)s0 * DH + half * 16;
    v16bf kf[4];
#pragma unroll
    for (int kc = 0; kc < 4; ++kc) kf[kc] = load_b16(kp + kc * 32);
    v8f sc = vzero();
#pragma unroll
    for (int kc = 0; kc < 4; ++kc) sc = wmma_bf16(qa[kc], kf[kc], sc);
#pragma unroll
    for (int r = 0; r < 8; ++r) ldsS[rg_s][ty_s][hs][r + 8 * half][ln] = sc[r];
    __syncthreads();

    // ---- phase 2: online softmax, waves 0-3 (one (rg,type) combo each) ----
    if (wave < 4) {
      const int rg = wave & 1, ty = wave >> 1;
      const int row = ln, hh = half;  // lane = (row, coltile)
      float v[16];
#pragma unroll
      for (int j = 0; j < 16; ++j) v[j] = ldsS[rg][ty][hh][row][j];
      float mx = v[0];
#pragma unroll
      for (int j = 1; j < 16; ++j) mx = fmaxf(mx, v[j]);
      mx = fmaxf(mx, __shfl_xor(mx, 16, 32));
      float mold = mstate[rg][ty][row];
      float mn = fmaxf(mold, mx);
      float al = __expf(mold - mn);
      float ssum = 0.f;
#pragma unroll
      for (int j = 0; j < 16; ++j) {
        float p = __expf(v[j] - mn);
        ssum += p;
        ldsP[rg][ty][row * 40 + hh * 16 + j] = (bf16_t)p;
      }
      ssum += __shfl_xor(ssum, 16, 32);
      if (hh == 0) {
        mstate[rg][ty][row] = mn;
        sstate[rg][ty][row] = sstate[rg][ty][row] * al + ssum;
        alphaL[rg][ty][row] = al;
      }
    }
    __syncthreads();

    // ---- phase 3: prefetch V fragments + P fragments, rescale, 6 PV WMMAs ----
    v16bf vf[2], vtf[2];
#pragma unroll
    for (int g = 0; g < 2; ++g) {
      int d = dq * 32 + g * 16 + ln;
      vf[g] = load_b16(vb + (size_t)d * S + s0 + half * 16);
      vtf[g] = load_b16(vtb + (size_t)d * S + s0 + half * 16);
    }
    v16bf pa = load_a16(&ldsP[rg_p][0][ln * 40], half);
    v16bf pta = load_a16(&ldsP[rg_p][1][ln * 40], half);
#pragma unroll
    for (int r = 0; r < 8; ++r) {
      float aR = alphaL[rg_p][0][r + 8 * half];
      float aT = alphaL[rg_p][1][r + 8 * half];
      accR[0][r] *= aR; accR[1][r] *= aR;
      accT[0][r] *= aT; accT[1][r] *= aT;
      accTV[0][r] *= aT; accTV[1][r] *= aT;
    }
#pragma unroll
    for (int g = 0; g < 2; ++g) {
      accR[g] = wmma_bf16(pa, vf[g], accR[g]);
      accTV[g] = wmma_bf16(pta, vf[g], accTV[g]);
      accT[g] = wmma_bf16(pta, vtf[g], accT[g]);
    }
    __syncthreads();
  }

  // ---- writeback (normalize by final sums) ----
#pragma unroll
  for (int g = 0; g < 2; ++g)
#pragma unroll
    for (int r = 0; r < 8; ++r) {
      int rrow = r + 8 * half;
      float sRr = sstate[rg_p][0][rrow];
      float sTr = sstate[rg_p][1][rrow];
      int row = trow0 + rg_p * 16 + rrow;
      int d = dq * 32 + g * 16 + ln;
      size_t off = ((size_t)bh * T + row) * DH + d;
      ctx[off] = accR[g][r] / sRr;
      ctxTV[off] = accTV[g][r] / sTr;
      ctxT[off] = accT[g][r] / sTr;
    }
}

// ---------------- gate: sigmoid([ctx|ctx_t|q] @ Wg + bg), block per (b,t) ----------------
__global__ __launch_bounds__(256) void gate_k(const float* __restrict__ ctx,
                                              const float* __restrict__ ctxT,
                                              const bf16_t* __restrict__ qh,
                                              const float* __restrict__ Wg,
                                              const float* __restrict__ bg,
                                              float* __restrict__ gate) {
  __shared__ float acc[8];
  const int bt = blockIdx.x;
  const int b = bt / T, t = bt % T;
  const int tid = threadIdx.x;
  if (tid < 8) acc[tid] = 0.f;
  __syncthreads();
  float part[8] = {0, 0, 0, 0, 0, 0, 0, 0};
  for (int row = tid; row < 3 * E; row += 256) {
    int h = row / (3 * DH);
    int rs = row % (3 * DH);
    size_t cb = (((size_t)b * H + h) * T + t) * DH;
    float f;
    if (rs < DH) f = ctx[cb + rs];
    else if (rs < 2 * DH) f = ctxT[cb + rs - DH];
    else f = (float)qh[cb + rs - 2 * DH];
#pragma unroll
    for (int j = 0; j < 8; ++j) part[j] += f * Wg[(size_t)row * 8 + j];
  }
#pragma unroll
  for (int j = 0; j < 8; ++j) atomicAdd(&acc[j], part[j]);
  __syncthreads();
  if (tid < 8) {
    float z = acc[tid] + bg[tid];
    gate[(size_t)bt * 8 + tid] = 1.f / (1.f + __expf(-z));
  }
}

// ---------------- mix: out_pre = g*ctx + (1-g)*ctx_tv -> bf16 [B*T, E] ----------------
__global__ void mix_k(const float* __restrict__ ctx, const float* __restrict__ ctxTV,
                      const float* __restrict__ gate, bf16_t* __restrict__ out) {
  size_t i = (size_t)blockIdx.x * 256 + threadIdx.x;
  if (i >= (size_t)Bb * H * T * DH) return;
  int d = (int)(i & 127);
  size_t r = i >> 7;
  int t = (int)(r % T);
  size_t r2 = r / T;
  int h = (int)(r2 % H);
  int b = (int)(r2 / H);
  float g = gate[((size_t)b * T + t) * 8 + h];
  float v = g * ctx[i] + (1.f - g) * ctxTV[i];
  out[((size_t)b * T + t) * E + h * DH + d] = (bf16_t)v;
}

extern "C" void kernel_launch(void* const* d_in, const int* in_sizes, int n_in,
                              void* d_out, int out_size, void* d_ws, size_t ws_size,
                              hipStream_t stream) {
  (void)in_sizes; (void)n_in; (void)out_size; (void)ws_size;
  const float* hs = (const float*)d_in[0];
  const float* kv = (const float*)d_in[1];
  const float* kvt = (const float*)d_in[2];
  const float* Wq = (const float*)d_in[3]; const float* bq = (const float*)d_in[4];
  const float* Wk = (const float*)d_in[5]; const float* bk = (const float*)d_in[6];
  const float* Wv = (const float*)d_in[7]; const float* bv = (const float*)d_in[8];
  const float* Wtk = (const float*)d_in[9]; const float* btk = (const float*)d_in[10];
  const float* Wtv = (const float*)d_in[11]; const float* btv = (const float*)d_in[12];
  const float* Wg = (const float*)d_in[13]; const float* bg = (const float*)d_in[14];
  const float* Wo = (const float*)d_in[15]; const float* bo = (const float*)d_in[16];
  float* out = (float*)d_out;

  const size_t nXh = (size_t)Bb * T * E;    // 1 Mi
  const size_t nXkv = (size_t)Bb * S * E;   // 8 Mi
  const size_t nXtp = (size_t)Bb * S * TP;  // 4 Mi
  const size_t nCtx = (size_t)Bb * H * T * DH;

  size_t o = 0;
  char* base = (char*)d_ws;
  auto take = [&](size_t bytes) -> void* {
    void* p = base + o;
    o += (bytes + 255) & ~(size_t)255;
    return p;
  };
  bf16_t* Xh = (bf16_t*)take(nXh * 2);
  bf16_t* Xkv = (bf16_t*)take(nXkv * 2);
  bf16_t* Xtp = (bf16_t*)take(nXtp * 2);
  bf16_t* WqT = (bf16_t*)take((size_t)E * E * 2);
  bf16_t* WkT = (bf16_t*)take((size_t)E * E * 2);
  bf16_t* WvT = (bf16_t*)take((size_t)E * E * 2);
  bf16_t* WoT = (bf16_t*)take((size_t)E * E * 2);
  bf16_t* WtkT = (bf16_t*)take((size_t)E * TP * 2);
  bf16_t* WtvT = (bf16_t*)take((size_t)E * TP * 2);
  bf16_t* qh = (bf16_t*)take(nXh * 2);
  bf16_t* kh = (bf16_t*)take(nXkv * 2);
  bf16_t* kth = (bf16_t*)take(nXkv * 2);
  bf16_t* vTb = (bf16_t*)take(nXkv * 2);
  bf16_t* vtTb = (bf16_t*)take(nXkv * 2);
  float* ctx = (float*)take(nCtx * 4);
  float* ctxT = (float*)take(nCtx * 4);
  float* ctxTV = (float*)take(nCtx * 4);
  float* gatep = (float*)take((size_t)Bb * T * H * 4);
  bf16_t* mixb = (bf16_t*)take(nXh * 2);

  // 1. convert activations
  cvt_k<<<(unsigned)((nXh + 255) / 256), 256, 0, stream>>>(hs, Xh, nXh);
  cvt_k<<<(unsigned)((nXkv + 255) / 256), 256, 0, stream>>>(kv, Xkv, nXkv);
  cvt_k<<<(unsigned)((nXtp + 255) / 256), 256, 0, stream>>>(kvt, Xtp, nXtp);
  // 2. transpose weights
  const unsigned gEE = (unsigned)(((size_t)E * E + 255) / 256);
  const unsigned gET = (unsigned)(((size_t)E * TP + 255) / 256);
  trans_k<<<gEE, 256, 0, stream>>>(Wq, WqT, E, E);
  trans_k<<<gEE, 256, 0, stream>>>(Wk, WkT, E, E);
  trans_k<<<gEE, 256, 0, stream>>>(Wv, WvT, E, E);
  trans_k<<<gEE, 256, 0, stream>>>(Wo, WoT, E, E);
  trans_k<<<gET, 256, 0, stream>>>(Wtk, WtkT, TP, E);
  trans_k<<<gET, 256, 0, stream>>>(Wtv, WtvT, TP, E);
  // 3. projections
  const float scale = 0.08838834764831845f;  // 1/sqrt(128)
  dim3 gq(E / 64, (Bb * T) / 64);
  dim3 gkv(E / 64, (Bb * S) / 64);
  gemm_k<1, true><<<gq, 128, 0, stream>>>(Xh, WqT, bq, qh, Bb * T, E, E, T, scale);
  gemm_k<1, true><<<gkv, 128, 0, stream>>>(Xkv, WkT, bk, kh, Bb * S, E, E, S, 1.f);
  gemm_k<2, true><<<gkv, 128, 0, stream>>>(Xkv, WvT, bv, vTb, Bb * S, E, E, S, 1.f);
  gemm_k<1, true><<<gkv, 128, 0, stream>>>(Xtp, WtkT, btk, kth, Bb * S, E, TP, S, 1.f);
  gemm_k<2, true><<<gkv, 128, 0, stream>>>(Xtp, WtvT, btv, vtTb, Bb * S, E, TP, S, 1.f);
  // 4. cooperative streaming attention -> three contexts
  attn_k<<<Bb * H * (T / 32), 256, 0, stream>>>(qh, kh, kth, vTb, vtTb, ctx, ctxT, ctxTV);
  // 5. gate
  gate_k<<<Bb * T, 256, 0, stream>>>(ctx, ctxT, qh, Wg, bg, gatep);
  // 6. mix
  mix_k<<<(unsigned)((nCtx + 255) / 256), 256, 0, stream>>>(ctx, ctxTV, gatep, mixb);
  // 7. output projection -> d_out (f32)
  gemm_k<0, false><<<gq, 128, 0, stream>>>(mixb, WoT, bo, out, Bb * T, E, E, T, 1.f);
}
